// SeparateWindowedAttention_43344809951957
// MI455X (gfx1250) — compile-verified, matrix operand order checked by reading, not run on hardware
//
#include <hip/hip_runtime.h>
#include <hip/hip_bf16.h>

// ---------------- WMMA types / helpers (CDNA5 wave32) ----------------
typedef __attribute__((ext_vector_type(16))) __bf16 v16bf;
typedef __attribute__((ext_vector_type(8)))  float  v8f;

__device__ inline unsigned short f2bf(float x) {
    union { float f; unsigned u; } c; c.f = x;
    unsigned r = c.u + 0x7FFFu + ((c.u >> 16) & 1u);   // RNE
    return (unsigned short)(r >> 16);
}

__device__ inline v8f wmma_bf16(v16bf a, v16bf b, v8f c) {
    return __builtin_amdgcn_wmma_f32_16x16x32_bf16(
        /*neg_a=*/false, a, /*neg_b=*/false, b,
        /*c_mod=*/(short)0, c, /*reuse_a=*/false, /*reuse_b=*/false);
}

// CDNA5 async global->LDS copy, 16 bytes per active lane (ASYNCcnt-tracked).
// GV addressing: 64-bit VGPR global address; VDST VGPR holds LDS byte offset
// (low 32 bits of the generic shared-space pointer == LDS offset).
__device__ inline void async_copy16(const void* g, void* lds) {
    unsigned loff = (unsigned)(size_t)lds;
    asm volatile("global_load_async_to_lds_b128 %0, %1, off"
                 :: "v"(loff), "v"(g) : "memory");
}
__device__ inline void wait_async0() {
    asm volatile("s_wait_asynccnt 0x0" ::: "memory");
}

// Load a 16x32 bf16 fragment from LDS laid out row-major [m][k] (pitch even).
// Same routine serves B fragments when LDS holds B transposed as [n][k].
__device__ inline v16bf load_frag(const unsigned short* base, int pitch,
                                  int mOff, int kOff) {
    int lane = threadIdx.x & 31;
    int m   = mOff + (lane & 15);
    int khi = (lane >> 4) << 3;     // 0 or 8
    union { v16bf v; unsigned u[8]; } f;
#pragma unroll
    for (int p = 0; p < 8; ++p) {
        int k = ((p < 4) ? (2 * p) : (2 * p + 8)) + khi + kOff;
        f.u[p] = *(const unsigned*)(base + m * pitch + k);  // 4B aligned (k even)
    }
    return f.v;
}

// ---------------- fp32 -> bf16 elementwise ----------------
__global__ __launch_bounds__(256) void cvt_f2bf_kernel(
    const float* __restrict__ X, unsigned short* __restrict__ Y, size_t n)
{
    size_t i = (size_t)blockIdx.x * blockDim.x + threadIdx.x;
    size_t stride = (size_t)gridDim.x * blockDim.x;
    for (; i < n; i += stride) Y[i] = f2bf(X[i]);
}

// ---------------- weight transpose+convert: Wt[n][k] = bf16(W[k][n]) --------
__global__ __launch_bounds__(256) void transpose_cvt_kernel(
    const float* __restrict__ W, unsigned short* __restrict__ Wt, int K, int N)
{
    size_t tot = (size_t)K * N;
    size_t stride = (size_t)gridDim.x * blockDim.x;
    for (size_t i = (size_t)blockIdx.x * blockDim.x + threadIdx.x; i < tot; i += stride) {
        size_t n = i / (size_t)K, k = i - n * (size_t)K;  // output-contiguous
        Wt[i] = f2bf(W[k * (size_t)N + n]);
    }
}

// ---------------- GEMM: C = A(bf16 MxK) @ Wt(bf16 NxK)^T + bias -------------
// Block 64(M) x 128(N), 256 threads = 8 waves; wave tile 16x64.
// Tiles staged with global_load_async_to_lds_b128 (no VGPR round trip).
template <int OBF>
__global__ __launch_bounds__(256) void gemm_bf16_kernel(
    const unsigned short* __restrict__ A,   // M x K bf16
    const unsigned short* __restrict__ Wt,  // N x K bf16 (pre-transposed)
    const float* __restrict__ bias,
    float* __restrict__ Cf, unsigned short* __restrict__ Cb,
    int M, int K, int N)
{
    __shared__ __align__(16) unsigned short As[64][32];    // [m][k]
    __shared__ __align__(16) unsigned short Bs[128][32];   // [n][k]
    int tid  = threadIdx.x;
    int wave = tid >> 5;
    int bm = blockIdx.y * 64;
    int bn = blockIdx.x * 128;
    int wm = (wave >> 1) * 16;
    int wn = (wave & 1) * 64;
    // per-thread 16B chunks: A = 256 chunks, B = 512 chunks
    int ar  = tid >> 2,        ac  = (tid & 3) * 8;
    int bn0 = (2 * tid) >> 2,  bk0 = ((2 * tid) & 3) * 8;
    int bn1 = (2 * tid + 1) >> 2, bk1 = ((2 * tid + 1) & 3) * 8;
    const unsigned short* aSrc = A  + (size_t)(bm + ar)  * K + ac;
    const unsigned short* bSrc0 = Wt + (size_t)(bn + bn0) * K + bk0;
    const unsigned short* bSrc1 = Wt + (size_t)(bn + bn1) * K + bk1;
    v8f acc[4] = {};

    for (int k0 = 0; k0 < K; k0 += 32) {
        async_copy16(aSrc  + k0, &As[ar][ac]);
        async_copy16(bSrc0 + k0, &Bs[bn0][bk0]);
        async_copy16(bSrc1 + k0, &Bs[bn1][bk1]);
        if (k0 + 32 < K) __builtin_prefetch(aSrc + k0 + 32, 0, 0);
        wait_async0();
        __syncthreads();
        v16bf af = load_frag(&As[0][0], 32, wm, 0);
#pragma unroll
        for (int t = 0; t < 4; ++t) {
            v16bf bf_ = load_frag(&Bs[0][0], 32, wn + t * 16, 0);
            acc[t] = wmma_bf16(af, bf_, acc[t]);
        }
        __syncthreads();
    }
    // epilogue: C/D layout — VGPR r: (m=r, lanes 0-15) / (m=r+8, lanes 16-31)
    int lane = tid & 31;
    int nl = lane & 15, mo = (lane >> 4) << 3;
#pragma unroll
    for (int t = 0; t < 4; ++t) {
        int n = bn + wn + t * 16 + nl;
        float bv = bias[n];
#pragma unroll
        for (int r = 0; r < 8; ++r) {
            int m = bm + wm + r + mo;
            float v = acc[t][r] + bv;
            if (OBF) Cb[(size_t)m * N + n] = f2bf(v);
            else     Cf[(size_t)m * N + n] = v;
        }
    }
}

// ---------------- RMSNorm over full row, split q|k, write bf16 in-place -----
__global__ __launch_bounds__(256) void rmsnorm_split_kernel(
    const float* __restrict__ X, const float* __restrict__ g,
    unsigned short* __restrict__ QK, int C)
{
    int row = blockIdx.x;
    const float* x = X + (size_t)row * C;
    float xs[4]; int cs[4]; int cnt = 0;
    float ss = 0.f;
    for (int c = threadIdx.x; c < C; c += 256) {
        float v = x[c];
        xs[cnt] = v; cs[cnt] = c; ++cnt;
        ss += v * v;
    }
#pragma unroll
    for (int off = 16; off; off >>= 1) ss += __shfl_down(ss, off);
    __shared__ float red[8];
    __shared__ float sscale;
    int wave = threadIdx.x >> 5, lane = threadIdx.x & 31;
    if (lane == 0) red[wave] = ss;
    __syncthreads();
    if (threadIdx.x == 0) {
        float t = 0.f;
#pragma unroll
        for (int i = 0; i < 8; ++i) t += red[i];
        sscale = rsqrtf(t / (float)C + 1e-6f);
    }
    __syncthreads();
    float scale = sscale;
    unsigned short* dst = QK + (size_t)row * 2 * C;
    for (int i = 0; i < cnt; ++i)
        dst[cs[i]] = f2bf(xs[i] * scale * g[cs[i]]);
}

// ---------------- Windowed shifted attention, one block per (window, head) --
#define ATT_SCALE 0.11180339887498948f   // 1/sqrt(64+16)

__global__ __launch_bounds__(128) void win_attn_kernel(
    const unsigned short* __restrict__ fqk,   // pitch 2048: q@0, k@512 (bf16)
    const unsigned short* __restrict__ cqk,   // pitch 512:  q@0, k@128 (bf16)
    const unsigned short* __restrict__ Vb,    // (B,128,128,512) bf16
    const float* __restrict__ pos_emb,        // 15x15 fp32
    unsigned short* __restrict__ O)           // (B,128,128,512) bf16, un-shifted
{
    int win  = blockIdx.x;          // 0..255 (Wh=16 x Ww=16)
    int head = blockIdx.y;          // 0..7
    int b    = blockIdx.z;          // 0..3
    int wx = win >> 4;
    int wy = win & 15;

    __shared__ __align__(16) unsigned short Qs[64][96];   // [query][dim 64f+16c+pad]
    __shared__ __align__(16) unsigned short Ks[64][96];
    __shared__ __align__(16) unsigned short Vst[64][66];  // [dim][token]
    __shared__ __align__(16) unsigned short Ps[64][64];   // probs bf16 [q][k]
    __shared__ float Ss[64][64];
    __shared__ float pe[225];

    int tid = threadIdx.x;
    auto rowOf = [&](int n) {   // token -> flat (b,gh,gw) row, shift folded in
        int i = n >> 3, j = n & 7;
        int gh = (wx * 8 + i + 4) & 127;
        int gw = (wy * 8 + j + 4) & 127;
        return (b * 128 + gh) * 128 + gw;
    };

    for (int i = tid; i < 225; i += 128) pe[i] = pos_emb[i];

    {   // stage Q,K: 2 threads per token, 48 dims each
        int token = tid >> 1;
        int half  = tid & 1;
        int row = rowOf(token);
        const unsigned short* fq = fqk + (size_t)row * 2048 + head * 64;
        const unsigned short* fk = fq + 512;
        const unsigned short* cq = cqk + (size_t)row * 512 + head * 16;
        const unsigned short* ck = cq + 128;
#pragma unroll
        for (int d0 = 0; d0 < 48; ++d0) {
            int d = half * 48 + d0;
            unsigned short qv, kv;
            if (d < 64)      { qv = fq[d];      kv = fk[d];      }
            else if (d < 80) { qv = cq[d - 64]; kv = ck[d - 64]; }
            else             { qv = 0;          kv = 0;          }
            Qs[token][d] = qv; Ks[token][d] = kv;
        }
    }
    // stage V transposed: Vst[dim][token]
    for (int idx = tid; idx < 64 * 64; idx += 128) {
        int token = idx >> 6, dim = idx & 63;
        Vst[dim][token] = Vb[(size_t)rowOf(token) * 512 + head * 64 + dim];
    }
    __syncthreads();

    // ---- scores: 3 K-steps of 32 over padded 96 dims (feature||coord)
    int wave = tid >> 5;
    int wm = wave * 16;
    v8f sacc[4] = {};
#pragma unroll
    for (int ks = 0; ks < 3; ++ks) {
        v16bf a = load_frag(&Qs[0][0], 96, wm, ks * 32);
#pragma unroll
        for (int t = 0; t < 4; ++t) {
            v16bf bb = load_frag(&Ks[0][0], 96, t * 16, ks * 32);
            sacc[t] = wmma_bf16(a, bb, sacc[t]);
        }
    }
    int lane = tid & 31;
    int nl = lane & 15, mo = (lane >> 4) << 3;
    bool lastWy = (wy == 15), lastWx = (wx == 15);
#pragma unroll
    for (int t = 0; t < 4; ++t) {
#pragma unroll
        for (int r = 0; r < 8; ++r) {
            int m = wm + r + mo;        // query token
            int n = t * 16 + nl;        // key token
            int qi = m >> 3, qj = m & 7, ki = n >> 3, kj = n & 7;
            float s = sacc[t][r] * ATT_SCALE + pe[(qi - ki + 7) * 15 + (qj - kj + 7)];
            if (lastWy && ((m >= 32) != (n >= 32))) s -= 1e30f;   // 'row' mask
            if (lastWx && ((qj >= 4) != (kj >= 4))) s -= 1e30f;   // 'col' mask
            Ss[m][n] = s;
        }
    }
    __syncthreads();

    // ---- softmax, one thread per query row
    if (tid < 64) {
        float mx = -3.0e38f;
#pragma unroll 4
        for (int n = 0; n < 64; ++n) mx = fmaxf(mx, Ss[tid][n]);
        float sum = 0.f;
#pragma unroll 4
        for (int n = 0; n < 64; ++n) {
            float e = __expf(Ss[tid][n] - mx);
            Ss[tid][n] = e; sum += e;
        }
        float inv = 1.f / sum;
#pragma unroll 4
        for (int n = 0; n < 64; ++n) Ps[tid][n] = f2bf(Ss[tid][n] * inv);
    }
    __syncthreads();

    // ---- O = P @ V : K = 64 tokens -> 2 K-steps
    v8f oacc[4] = {};
#pragma unroll
    for (int ks = 0; ks < 2; ++ks) {
        v16bf a = load_frag(&Ps[0][0], 64, wm, ks * 32);
#pragma unroll
        for (int t = 0; t < 4; ++t) {
            v16bf bb = load_frag(&Vst[0][0], 66, t * 16, ks * 32);
            oacc[t] = wmma_bf16(a, bb, oacc[t]);
        }
    }
#pragma unroll
    for (int t = 0; t < 4; ++t) {
#pragma unroll
        for (int r = 0; r < 8; ++r) {
            int m = wm + r + mo;
            int d = t * 16 + nl;
            O[(size_t)rowOf(m) * 512 + head * 64 + d] = f2bf(oacc[t][r]);
        }
    }
}

// ---------------- Host-side orchestration ----------------
extern "C" void kernel_launch(void* const* d_in, const int* in_sizes, int n_in,
                              void* d_out, int out_size, void* d_ws, size_t ws_size,
                              hipStream_t stream) {
    (void)in_sizes; (void)n_in; (void)out_size; (void)ws_size;
    const float* features = (const float*)d_in[0];   // (4,128,128,512)
    const float* coords   = (const float*)d_in[1];   // (4,128,128,128)
    const float* w_fqk    = (const float*)d_in[2];   // (512,1024)
    const float* b_fqk    = (const float*)d_in[3];
    const float* g_fqk    = (const float*)d_in[4];
    const float* w_cqk    = (const float*)d_in[5];   // (128,256)
    const float* b_cqk    = (const float*)d_in[6];
    const float* g_cqk    = (const float*)d_in[7];
    const float* w_v      = (const float*)d_in[8];   // (512,512)
    const float* b_v      = (const float*)d_in[9];
    const float* w_out    = (const float*)d_in[10];  // (512,512)
    const float* b_out    = (const float*)d_in[11];
    const float* pos_emb  = (const float*)d_in[12];  // (15,15)
    float* out = (float*)d_out;                      // (4,128,128,512)

    const int M = 4 * 128 * 128;   // 65536 tokens
    char* ws = (char*)d_ws;
    size_t off = 0;
    float* fqk_raw = (float*)(ws + off);            off += (size_t)M * 1024 * 4;
    float* cqk_raw = (float*)(ws + off);            off += (size_t)M * 256 * 4;
    unsigned short* feat_bf = (unsigned short*)(ws + off); off += (size_t)M * 512 * 2;
    unsigned short* crd_bf  = (unsigned short*)(ws + off); off += (size_t)M * 128 * 2;
    unsigned short* fv_bf   = (unsigned short*)(ws + off); off += (size_t)M * 512 * 2;
    unsigned short* attn_bf = (unsigned short*)(ws + off); off += (size_t)M * 512 * 2;
    unsigned short* wt_fqk  = (unsigned short*)(ws + off); off += (size_t)1024 * 512 * 2;
    unsigned short* wt_cqk  = (unsigned short*)(ws + off); off += (size_t)256 * 128 * 2;
    unsigned short* wt_v    = (unsigned short*)(ws + off); off += (size_t)512 * 512 * 2;
    unsigned short* wt_out  = (unsigned short*)(ws + off);

    // 0) precision prep: activations -> bf16; weights -> bf16 transposed (N,K)
    cvt_f2bf_kernel<<<4096, 256, 0, stream>>>(features, feat_bf, (size_t)M * 512);
    cvt_f2bf_kernel<<<1024, 256, 0, stream>>>(coords, crd_bf, (size_t)M * 128);
    transpose_cvt_kernel<<<512, 256, 0, stream>>>(w_fqk, wt_fqk, 512, 1024);
    transpose_cvt_kernel<<<64,  256, 0, stream>>>(w_cqk, wt_cqk, 128, 256);
    transpose_cvt_kernel<<<256, 256, 0, stream>>>(w_v,   wt_v,   512, 512);
    transpose_cvt_kernel<<<256, 256, 0, stream>>>(w_out, wt_out, 512, 512);

    // 1) projections (bias fused); async-LDS staged bf16 WMMA GEMMs
    gemm_bf16_kernel<0><<<dim3(1024 / 128, M / 64), 256, 0, stream>>>(
        feat_bf, wt_fqk, b_fqk, fqk_raw, nullptr, M, 512, 1024);
    gemm_bf16_kernel<0><<<dim3(256 / 128, M / 64), 256, 0, stream>>>(
        crd_bf, wt_cqk, b_cqk, cqk_raw, nullptr, M, 128, 256);
    gemm_bf16_kernel<1><<<dim3(512 / 128, M / 64), 256, 0, stream>>>(
        feat_bf, wt_v, b_v, nullptr, fv_bf, M, 512, 512);

    // 2) rmsnorm over full qk rows; write bf16 q|k packed in-place
    rmsnorm_split_kernel<<<M, 256, 0, stream>>>(fqk_raw, g_fqk,
        (unsigned short*)fqk_raw, 1024);
    rmsnorm_split_kernel<<<M, 256, 0, stream>>>(cqk_raw, g_cqk,
        (unsigned short*)cqk_raw, 256);

    // 3) shifted windowed attention (shift folded into index math)
    win_attn_kernel<<<dim3(256, 8, 4), 128, 0, stream>>>(
        (const unsigned short*)fqk_raw, (const unsigned short*)cqk_raw,
        fv_bf, pos_emb, attn_bf);

    // 4) output projection -> fp32 result
    gemm_bf16_kernel<0><<<dim3(512 / 128, M / 64), 256, 0, stream>>>(
        attn_bf, wt_out, b_out, out, nullptr, M, 512, 512);
}
